// StyleHyperMixerBlock_3186865733837
// MI455X (gfx1250) — matrix operand order repaired and checked by reading, not compiled, not run on hardware
//
#include <hip/hip_runtime.h>
#include <math.h>

typedef __attribute__((ext_vector_type(16))) __bf16 v16bf;
typedef __attribute__((ext_vector_type(8)))  float  v8f;
typedef unsigned int u32x4v __attribute__((ext_vector_type(4)));
typedef int          i32x4v __attribute__((ext_vector_type(4)));
typedef int          i32x8v __attribute__((ext_vector_type(8)));

#define Bsz  8
#define Tsz  2048
#define Csz  768
#define WDsz 512
#define TDsz 384
#define CDsz 3072
#define Msz  (Bsz * Tsz)   /* 16384 */

// ---- tiles: 128x128 block, BK=64, 8 waves (4x2), 32x64 per wave, double-buffered LDS
#define BM 128
#define BN 128
#define BKt 64
#define LDP 72            /* BKt + 8 pad, bf16 units; row pitch 144B */

// ---------- TDM availability ----------
#if defined(__has_builtin)
#  if __has_builtin(__builtin_amdgcn_tensor_load_to_lds)
#    define HAVE_TDM 1
#  endif
#endif
#ifndef HAVE_TDM
#  define HAVE_TDM 0
#endif
#if __has_include(<hip/amd_detail/amd_gfx1250_TDM.h>)
#  define TDM_6ARG 1
#else
#  define TDM_6ARG 0
#endif

#if HAVE_TDM
#  if TDM_6ARG
#    pragma message("CDNA5: tensor_load_to_lds available (6-arg form)")
#  else
#    pragma message("CDNA5: tensor_load_to_lds available (5-arg form)")
#  endif
#else
#  pragma message("CDNA5: tensor_load_to_lds NOT available - manual A staging")
#endif

__device__ __forceinline__ void wait_tensorcnt0() {
#if defined(__has_builtin) && __has_builtin(__builtin_amdgcn_s_wait_tensorcnt)
    __builtin_amdgcn_s_wait_tensorcnt(0);
#else
    asm volatile("s_wait_tensorcnt 0x0" ::: "memory");
#endif
}

#if HAVE_TDM
// Issue a TDM 2D tile load: tile = BKt (dim0, contiguous) x BM (dim1 rows),
// bf16 elements, global row stride = lda elements, LDS padded to LDP pitch
// via D# pad fields (pad 16B after every 128B row).
__device__ __forceinline__ void tdm_issue_A(const __bf16* gsrc, int lda, unsigned lds_off)
{
    const unsigned long long ga = (unsigned long long)(__SIZE_TYPE__)gsrc;
    union { unsigned long long q[2]; u32x4v v; } g0;
    g0.q[0] = 1ull /*count=1*/ | ((unsigned long long)lds_off << 32);
    g0.q[1] = (ga & ((1ull << 57) - 1)) | (2ull << 62) /*type=image*/;

    // group1: data_size=2B, pad_enable, pad_interval=4 (128B), pad_amount=3 (16B)
    const unsigned w0 = (1u << 16) | (1u << 20) | (4u << 22) | (3u << 25);
    union { unsigned long long q[4]; i32x8v v; } g1;
    g1.q[0] = (unsigned long long)w0;                       // tensor_dim0 low16 = 0 (1M)
    g1.q[1] = 0x10ull                                        // tensor_dim0 = 1<<20 (hi16)
            | (0x100000ull << 16)                            // tensor_dim1 = 1<<20
            | ((unsigned long long)BKt << 48);               // tile_dim0 = 64
    g1.q[2] = (unsigned long long)BM                         // tile_dim1 = 128
            | ((unsigned long long)(unsigned)lda << 32);     // dim0_stride low32
    g1.q[3] = 0;                                             // stride hi16, dim1_stride
    const i32x4v z4 = {0, 0, 0, 0};
#if TDM_6ARG
    const i32x8v z8 = {0, 0, 0, 0, 0, 0, 0, 0};
    __builtin_amdgcn_tensor_load_to_lds(g0.v, g1.v, z4, z4, z8, 0);
#else
    __builtin_amdgcn_tensor_load_to_lds(g0.v, g1.v, z4, z4, 0);
#endif
}
#endif

// ---------- staging helpers ----------
__device__ __forceinline__ void loadB_regs(const __bf16* Bm, int ldb, int col0, int kk,
                                           int tid, uint4* r)
{   // 64k x 128n tile, 32 elems/thread: k = tid>>2, n0 = (tid&3)*32
    const int k = tid >> 2, n0 = (tid & 3) << 5;
    const uint4* s = (const uint4*)(Bm + (long)(kk + k) * ldb + col0 + n0);
    r[0] = s[0]; r[1] = s[1]; r[2] = s[2]; r[3] = s[3];
}
__device__ __forceinline__ void storeB_lds(__bf16* Bs, int tid, const uint4* r)
{   // transpose scatter: Bs[n][k]
    const int k = tid >> 2, n0 = (tid & 3) << 5;
    #pragma unroll
    for (int j = 0; j < 4; ++j) {
        const __bf16* e = (const __bf16*)&r[j];
        #pragma unroll
        for (int i = 0; i < 8; ++i) Bs[(n0 + j * 8 + i) * LDP + k] = e[i];
    }
}
__device__ __forceinline__ void loadA_regs(const __bf16* A, int lda, int row0, int kk,
                                           int tid, uint4* r)
{   // 128m x 64k tile, 32 elems/thread: m = tid>>1, k0 = (tid&1)*32
    const int m = tid >> 1, k0 = (tid & 1) << 5;
    const uint4* s = (const uint4*)(A + (long)(row0 + m) * lda + kk + k0);
    r[0] = s[0]; r[1] = s[1]; r[2] = s[2]; r[3] = s[3];
}
__device__ __forceinline__ void storeA_lds(__bf16* As, int tid, const uint4* r)
{
    const int m = tid >> 1, k0 = (tid & 1) << 5;
    uint4* d = (uint4*)(&As[m * LDP + k0]);
    d[0] = r[0]; d[1] = r[1]; d[2] = r[2]; d[3] = r[3];
}
__device__ __forceinline__ void loadAT_regs(const __bf16* A, int lda, int row0, int kk,
                                            int tid, uint4* r)
{   // logical A[m][k] = G[k][m]: k = tid>>2, m0 = (tid&3)*32
    const int k = tid >> 2, m0 = (tid & 3) << 5;
    const uint4* s = (const uint4*)(A + (long)(kk + k) * lda + row0 + m0);
    r[0] = s[0]; r[1] = s[1]; r[2] = s[2]; r[3] = s[3];
}
__device__ __forceinline__ void storeAT_lds(__bf16* As, int tid, const uint4* r)
{
    const int k = tid >> 2, m0 = (tid & 3) << 5;
    #pragma unroll
    for (int j = 0; j < 4; ++j) {
        const __bf16* e = (const __bf16*)&r[j];
        #pragma unroll
        for (int i = 0; i < 8; ++i) As[(m0 + j * 8 + i) * LDP + k] = e[i];
    }
}

// ---------------- tiled WMMA GEMM: C = op(A)*B (+bias) (+addSrc) ----------------
template <int TRANSA, int USE_TDM>
__global__ __launch_bounds__(256)
void k_gemm_bf16(const __bf16* __restrict__ A, const __bf16* __restrict__ Bm,
                 const float* __restrict__ bias, const float* __restrict__ addSrc,
                 float* __restrict__ outF, __bf16* __restrict__ outH,
                 int N, int K, int lda, int ldb, int ldc,
                 long strideA, long strideB, long strideC)
{
    __shared__ __align__(16) __bf16 As[2 * BM * LDP];
    __shared__ __align__(16) __bf16 Bs[2 * BN * LDP];

    const int tid   = threadIdx.x;
    const int lane  = tid & 31;
    const int wid   = tid >> 5;
    const int waveM = wid & 3;   // 4 waves x 32 rows = 128
    const int waveN = wid >> 2;  // 2 waves x 64 cols = 128
    const int half  = lane >> 4;
    const int l15   = lane & 15;

    const int row0 = blockIdx.y * BM;
    const int col0 = blockIdx.x * BN;
    A  += (long)blockIdx.z * strideA;
    Bm += (long)blockIdx.z * strideB;
    const long cbase = (long)blockIdx.z * strideC;

    v8f acc[2][4] = {};

    // ---- prologue: stage k-block 0 into buffer 0 ----
    {
        uint4 br[4];
        loadB_regs(Bm, ldb, col0, 0, tid, br);
        storeB_lds(Bs, tid, br);
#if HAVE_TDM
        if (USE_TDM) {
            if (wid == 0) {
                tdm_issue_A(A + (long)row0 * lda, lda,
                            (unsigned)(__SIZE_TYPE__)(&As[0]));
                wait_tensorcnt0();
            }
        } else
#endif
        {
            uint4 ar[4];
            if (TRANSA) { loadAT_regs(A, lda, row0, 0, tid, ar); storeAT_lds(As, tid, ar); }
            else        { loadA_regs (A, lda, row0, 0, tid, ar); storeA_lds (As, tid, ar); }
        }
        __syncthreads();
    }

    const int nStages = K / BKt;
    int buf = 0;
    for (int s = 0; s < nStages; ++s) {
        const int nxt = buf ^ 1;
        const int haveNext = (s + 1 < nStages);
        __bf16* Asn = As + nxt * (BM * LDP);
        __bf16* Bsn = Bs + nxt * (BN * LDP);

        // prefetch next tiles: B -> regs now, A via TDM (async) or regs
        uint4 br[4], ar[4];
        if (haveNext) {
            loadB_regs(Bm, ldb, col0, (s + 1) * BKt, tid, br);
#if HAVE_TDM
            if (USE_TDM) {
                if (wid == 0)
                    tdm_issue_A(A + (long)row0 * lda + (s + 1) * BKt, lda,
                                (unsigned)(__SIZE_TYPE__)Asn);
            } else
#endif
            {
                if (TRANSA) loadAT_regs(A, lda, row0, (s + 1) * BKt, tid, ar);
                else        loadA_regs (A, lda, row0, (s + 1) * BKt, tid, ar);
            }
        }

        // ---- compute on current buffer: 2 k-steps x (2x4) WMMA ----
        const __bf16* Asb = As + buf * (BM * LDP);
        const __bf16* Bsb = Bs + buf * (BN * LDP);
        #pragma unroll
        for (int ks = 0; ks < 2; ++ks) {
            union Frag { uint4 q[2]; v16bf v; };
            Frag a[2], bb[4];
            #pragma unroll
            for (int fm = 0; fm < 2; ++fm) {
                const __bf16* rp = &Asb[(waveM * 32 + fm * 16 + l15) * LDP + ks * 32];
                a[fm].q[0] = *(const uint4*)(rp + half * 8);
                a[fm].q[1] = *(const uint4*)(rp + 16 + half * 8);
            }
            #pragma unroll
            for (int fn = 0; fn < 4; ++fn) {
                const __bf16* rp = &Bsb[(waveN * 64 + fn * 16 + l15) * LDP + ks * 32];
                bb[fn].q[0] = *(const uint4*)(rp + half * 8);
                bb[fn].q[1] = *(const uint4*)(rp + 16 + half * 8);
            }
            #pragma unroll
            for (int fm = 0; fm < 2; ++fm)
                #pragma unroll
                for (int fn = 0; fn < 4; ++fn)
                    acc[fm][fn] = __builtin_amdgcn_wmma_f32_16x16x32_bf16(
                        false, a[fm].v, false, bb[fn].v, (short)0, acc[fm][fn],
                        false, false);
        }

        // ---- drain prefetch into next buffer ----
        if (haveNext) {
            storeB_lds(Bsn, tid, br);
#if HAVE_TDM
            if (USE_TDM) {
                if (wid == 0) wait_tensorcnt0();
            } else
#endif
            {
                if (TRANSA) storeAT_lds(Asn, tid, ar);
                else        storeA_lds (Asn, tid, ar);
            }
        }
        __syncthreads();
        buf = nxt;
    }

    // ---- epilogue: D layout: vgpr r -> M = r + 8*half, N = l15 ----
    const int mb = row0 + waveM * 32;
    const int nb = col0 + waveN * 64;
    #pragma unroll
    for (int fm = 0; fm < 2; ++fm) {
        #pragma unroll
        for (int fn = 0; fn < 4; ++fn) {
            #pragma unroll
            for (int r = 0; r < 8; ++r) {
                const int m = mb + fm * 16 + half * 8 + r;
                const int n = nb + fn * 16 + l15;
                const long off = cbase + (long)m * ldc + n;
                float v = acc[fm][fn][r];
                if (bias)   v += bias[n];
                if (addSrc) v += addSrc[off];
                if (outF)   outF[off] = v;
                if (outH)   outH[off] = (__bf16)v;
            }
        }
    }
}

// ---------------- LayerNorm over C=768, emits fp32 and/or bf16 ----------------
__global__ __launch_bounds__(256)
void k_layernorm(const float* __restrict__ x, const float* __restrict__ g,
                 const float* __restrict__ beta, float* __restrict__ outF,
                 __bf16* __restrict__ outH)
{
    const long row = blockIdx.x;
    const float* xr = x + row * Csz;
    float v[3];
    float s = 0.f, ss = 0.f;
    #pragma unroll
    for (int i = 0; i < 3; ++i) {
        v[i] = xr[threadIdx.x + i * 256];
        s += v[i]; ss += v[i] * v[i];
    }
    #pragma unroll
    for (int o = 16; o > 0; o >>= 1) { s += __shfl_xor(s, o); ss += __shfl_xor(ss, o); }
    __shared__ float ps[8], pss[8], stats[2];
    const int lane = threadIdx.x & 31, wid = threadIdx.x >> 5;
    if (lane == 0) { ps[wid] = s; pss[wid] = ss; }
    __syncthreads();
    if (threadIdx.x == 0) {
        float ts = 0.f, tss = 0.f;
        for (int i = 0; i < 8; ++i) { ts += ps[i]; tss += pss[i]; }
        const float mu = ts / Csz;
        stats[0] = mu;
        stats[1] = rsqrtf(tss / Csz - mu * mu + 1e-6f);
    }
    __syncthreads();
    const float mu = stats[0], rs = stats[1];
    #pragma unroll
    for (int i = 0; i < 3; ++i) {
        const int c = threadIdx.x + i * 256;
        const float y = (v[i] - mu) * rs * g[c] + beta[c];
        if (outF) outF[row * Csz + c] = y;
        if (outH) outH[row * Csz + c] = (__bf16)y;
    }
}

// ---------------- small style GEMV: out[b,h] = w[b,sel] @ W + bias ----------------
__global__ void k_stylevec(const float* __restrict__ w, int sel,
                           const float* __restrict__ W, const float* __restrict__ bias,
                           float* __restrict__ out, int H)
{
    const int id = blockIdx.x * 256 + threadIdx.x;
    if (id >= Bsz * H) return;
    const int b = id / H, h = id - b * H;
    const float* wr = w + (long)b * (2 * WDsz) + sel * WDsz;
    float acc = bias[h];
    for (int k = 0; k < WDsz; ++k) acc += wr[k] * W[(long)k * H + h];
    out[id] = acc;
}

// ---------------- GLU gate + optional per-batch style scale ----------------
__global__ void k_glu(const __bf16* __restrict__ h, const float* __restrict__ svec,
                      __bf16* __restrict__ out, int H)
{
    const long id = (long)blockIdx.x * 256 + threadIdx.x;
    if (id >= (long)Msz * H) return;
    const long r = id / H;
    const int  j = (int)(id - r * H);
    const int  b = (int)(r / Tsz);
    const __bf16* hr = h + r * (2L * H);
    const float a = (float)hr[j];
    const float g = (float)hr[H + j];
    float val = a / (1.f + __expf(-g));
    if (svec) val *= svec[(long)b * H + j];
    out[id] = (__bf16)val;
}

// ---------------- exact GELU + bf16 convert ----------------
__global__ void k_gelu_cvt(const float* __restrict__ in, __bf16* __restrict__ out, long n)
{
    const long id = (long)blockIdx.x * 256 + threadIdx.x;
    if (id >= n) return;
    const float x = in[id];
    out[id] = (__bf16)(0.5f * x * (1.f + erff(x * 0.70710678f)));
}

__global__ void k_cvt_bf16(const float* __restrict__ in, __bf16* __restrict__ out, long n)
{
    const long id = (long)blockIdx.x * 256 + threadIdx.x;
    if (id < n) out[id] = (__bf16)in[id];
}

// =====================================================================
extern "C" void kernel_launch(void* const* d_in, const int* in_sizes, int n_in,
                              void* d_out, int out_size, void* d_ws, size_t ws_size,
                              hipStream_t stream)
{
    const float* x    = (const float*)d_in[0];
    const float* w    = (const float*)d_in[1];
    const float* ln1g = (const float*)d_in[2];
    const float* ln1b = (const float*)d_in[3];
    const float* ln2g = (const float*)d_in[4];
    const float* ln2b = (const float*)d_in[5];
    const float* g1W  = (const float*)d_in[6];
    const float* g1b  = (const float*)d_in[7];
    const float* s1W  = (const float*)d_in[8];
    const float* s1b  = (const float*)d_in[9];
    const float* m1W  = (const float*)d_in[10];
    const float* m1b  = (const float*)d_in[11];
    const float* g2W  = (const float*)d_in[12];
    const float* g2b  = (const float*)d_in[13];
    const float* s2W  = (const float*)d_in[14];
    const float* s2b  = (const float*)d_in[15];
    const float* m2W  = (const float*)d_in[16];
    const float* m2b  = (const float*)d_in[17];
    const float* gsW  = (const float*)d_in[18];
    const float* gsb  = (const float*)d_in[19];
    const float* ssW  = (const float*)d_in[20];
    const float* ssb  = (const float*)d_in[21];
    const float* msW  = (const float*)d_in[22];
    const float* msb  = (const float*)d_in[23];
    const float* r1W  = (const float*)d_in[24];
    const float* r1b  = (const float*)d_in[25];
    const float* r2W  = (const float*)d_in[26];
    const float* r2b  = (const float*)d_in[27];
    float* out = (float*)d_out;

    // ---- workspace layout ----
    char* cur = (char*)d_ws;
    auto alloc = [&](long bytes) -> char* {
        char* p = cur; cur += (bytes + 255) & ~255L; return p;
    };
    float*  xres  = (float*) alloc((long)Msz * Csz * 4);
    float*  mixf  = (float*) alloc((long)Bsz * TDsz * Csz * 4);
    float*  s1    = (float*) alloc((long)Bsz * TDsz * 4);
    float*  s2    = (float*) alloc((long)Bsz * TDsz * 4);
    float*  ssv   = (float*) alloc((long)Bsz * (CDsz/2) * 4);
    __bf16* xb    = (__bf16*)alloc((long)Msz * Csz * 2);
    __bf16* hbuf  = (__bf16*)alloc((long)Msz * CDsz * 2);
    __bf16* abuf  = (__bf16*)alloc((long)Msz * (CDsz/2) * 2);
    __bf16* t1    = (__bf16*)alloc((long)Msz * TDsz * 2);
    __bf16* t2    = (__bf16*)alloc((long)Msz * TDsz * 2);
    __bf16* mixh  = (__bf16*)alloc((long)Bsz * TDsz * Csz * 2);
    __bf16* wg1   = (__bf16*)alloc((long)Csz * Csz * 2);
    __bf16* wg2   = (__bf16*)alloc((long)Csz * Csz * 2);
    __bf16* wm1   = (__bf16*)alloc((long)TDsz * TDsz * 2);
    __bf16* wm2   = (__bf16*)alloc((long)TDsz * TDsz * 2);
    __bf16* wgs   = (__bf16*)alloc((long)Csz * CDsz * 2);
    __bf16* wms   = (__bf16*)alloc((long)(CDsz/2) * Csz * 2);
    __bf16* wr1   = (__bf16*)alloc((long)Csz * Csz * 2);
    __bf16* wr2   = (__bf16*)alloc((long)TDsz * Csz * 2);

    auto cvt = [&](const float* src, __bf16* dst, long n) {
        k_cvt_bf16<<<dim3((unsigned)((n + 255) / 256)), dim3(256), 0, stream>>>(src, dst, n);
    };
    auto gemm = [&](const __bf16* A, const __bf16* Bm, const float* bias,
                    const float* add, float* oF, __bf16* oH,
                    int M, int N, int K, int lda, int ldb, int ldc, int transA,
                    long sA, long sB, long sC, int batch) {
        dim3 grid(N / BN, M / BM, batch);
        if (transA) {
            k_gemm_bf16<1, 0><<<grid, dim3(256), 0, stream>>>(
                A, Bm, bias, add, oF, oH, N, K, lda, ldb, ldc, sA, sB, sC);
        } else {
#if HAVE_TDM
            k_gemm_bf16<0, 1><<<grid, dim3(256), 0, stream>>>(
                A, Bm, bias, add, oF, oH, N, K, lda, ldb, ldc, sA, sB, sC);
#else
            k_gemm_bf16<0, 0><<<grid, dim3(256), 0, stream>>>(
                A, Bm, bias, add, oF, oH, N, K, lda, ldb, ldc, sA, sB, sC);
#endif
        }
    };
    auto glu = [&](const __bf16* h, const float* sv, __bf16* o, int H) {
        long n = (long)Msz * H;
        k_glu<<<dim3((unsigned)((n + 255) / 256)), dim3(256), 0, stream>>>(h, sv, o, H);
    };

    // 0) weights -> bf16
    cvt(g1W, wg1, (long)Csz * Csz);
    cvt(g2W, wg2, (long)Csz * Csz);
    cvt(m1W, wm1, (long)TDsz * TDsz);
    cvt(m2W, wm2, (long)TDsz * TDsz);
    cvt(gsW, wgs, (long)Csz * CDsz);
    cvt(msW, wms, (long)(CDsz/2) * Csz);
    cvt(r1W, wr1, (long)Csz * Csz);
    cvt(r2W, wr2, (long)TDsz * Csz);

    // 0b) LN1: x -> xres(f32) + xb(bf16)
    k_layernorm<<<dim3(Msz), dim3(256), 0, stream>>>(x, ln1g, ln1b, xres, xb);

    // 0c) style vectors
    k_stylevec<<<dim3((Bsz*TDsz + 255)/256), dim3(256), 0, stream>>>(w, 0, s1W, s1b, s1, TDsz);
    k_stylevec<<<dim3((Bsz*TDsz + 255)/256), dim3(256), 0, stream>>>(w, 0, s2W, s2b, s2, TDsz);
    k_stylevec<<<dim3((Bsz*(CDsz/2) + 255)/256), dim3(256), 0, stream>>>(w, 1, ssW, ssb, ssv, CDsz/2);

    // 1) h1 = xln @ g1W + g1b ; a1 = glu(h1)*s1 ; t1 = a1 @ m1W + m1b
    gemm(xb, wg1, g1b, nullptr, nullptr, hbuf, Msz, Csz, Csz, Csz, Csz, Csz, 0, 0, 0, 0, 1);
    glu(hbuf, s1, abuf, TDsz);
    gemm(abuf, wm1, m1b, nullptr, nullptr, t1, Msz, TDsz, TDsz, TDsz, TDsz, TDsz, 0, 0, 0, 0, 1);

    // 2) same for h2 -> t2
    gemm(xb, wg2, g2b, nullptr, nullptr, hbuf, Msz, Csz, Csz, Csz, Csz, Csz, 0, 0, 0, 0, 1);
    glu(hbuf, s2, abuf, TDsz);
    gemm(abuf, wm2, m2b, nullptr, nullptr, t2, Msz, TDsz, TDsz, TDsz, TDsz, TDsz, 0, 0, 0, 0, 1);

    // 3) mix[b] = t1[b]^T @ xln[b]  (384 x 768, K = 2048), then exact GELU -> bf16
    gemm(t1, xb, nullptr, nullptr, mixf, nullptr, TDsz, Csz, Tsz, TDsz, Csz, Csz, 1,
         (long)Tsz * TDsz, (long)Tsz * Csz, (long)TDsz * Csz, Bsz);
    {
        long n = (long)Bsz * TDsz * Csz;
        k_gelu_cvt<<<dim3((unsigned)((n + 255) / 256)), dim3(256), 0, stream>>>(mixf, mixh, n);
    }

    // 4) x1 = xln + t2[b] @ mix[b]   (in-place residual into xres)
    gemm(t2, mixh, nullptr, xres, xres, nullptr, Tsz, Csz, TDsz, TDsz, Csz, Csz, 0,
         (long)Tsz * TDsz, (long)TDsz * Csz, (long)Tsz * Csz, Bsz);

    // 5) LN2(x1) -> xb ; hs = xb @ gsW + gsb ; as = glu(hs)*ss
    k_layernorm<<<dim3(Msz), dim3(256), 0, stream>>>(xres, ln2g, ln2b, nullptr, xb);
    gemm(xb, wgs, gsb, nullptr, nullptr, hbuf, Msz, CDsz, Csz, Csz, CDsz, CDsz, 0, 0, 0, 0, 1);
    glu(hbuf, ssv, abuf, CDsz/2);

    // 6) x2 = x1 + as @ msW + msb  (fp32 in-place into xres, bf16 copy into xb)
    gemm(abuf, wms, msb, xres, xres, xb, Msz, Csz, CDsz/2, CDsz/2, Csz, Csz, 0, 0, 0, 0, 1);

    // 7) hr = x2 @ r1W + r1b ; ar = glu(hr) ; out = ar @ r2W + r2b
    gemm(xb, wr1, r1b, nullptr, nullptr, hbuf, Msz, Csz, Csz, Csz, Csz, Csz, 0, 0, 0, 0, 1);
    glu(hbuf, nullptr, abuf, TDsz);
    gemm(abuf, wr2, r2b, nullptr, out, nullptr, Msz, Csz, TDsz, TDsz, Csz, Csz, 0, 0, 0, 0, 1);

    (void)in_sizes; (void)n_in; (void)out_size; (void)ws_size;
}
